// Test_65249143160991
// MI455X (gfx1250) — matrix operand order repaired and checked by reading, not compiled, not run on hardware
//
#include <hip/hip_runtime.h>
#include <hip/hip_bf16.h>

typedef __attribute__((ext_vector_type(16))) _Float16 v16h;
typedef __attribute__((ext_vector_type(8)))  float    v8f;

#define N_NODES 500000
#define N_EDGES 1000000
#define ND 13
#define ED 5
#define KNB 4

// Branchless SELU: sc*max(x,0) + sc*al*(exp(min(x,0)) - 1).
// No EXEC-mask divergence; exp is a TRANS op that co-executes with WMMA.
__device__ __forceinline__ float selu_f(float x) {
  const float sc = 1.0507009873554805f;
  const float scal = 1.0507009873554805f * 1.6732632423543772f;
  float xp = fmaxf(x, 0.f);
  float xn = fminf(x, 0.f);
  return fmaf(sc, xp, scal * (__expf(xn) - 1.f));
}

// ---------------------------------------------------------------------------
// Tile GEMM: C[16 x NT*16] = A[16 x KT*32] * W + b, optional SELU.
// A lives in LDS, f16, row stride aStride halves (multiple of 32 -> every
// fragment sub-load is a 16B-aligned ds_load_b128).
// W is pre-packed per-(kt,nt) fragment in a contiguous-k B layout (A is read in the same order):
// lane l covers column n = n0+(l&15), K-half (l>>4)*16; its 16 halves are
// contiguous (32 bytes -> two global_load_b128).
// C/D layout: lane l covers col n=(l&15); VGPR r covers row (l>>4)*8 + r.
// ---------------------------------------------------------------------------
__device__ __forceinline__ void gemm_tile(
    const _Float16* A, int aStride,
    const _Float16* Wp, const float* bias, int fout,
    int KT, int NT,
    _Float16* Out, int oStride, int storeLimit, bool act)
{
  const int lane = threadIdx.x & 31;
  const int m    = lane & 15;
  const int hi   = lane >> 4;
  for (int nt = 0; nt < NT; ++nt) {
    v8f acc = {};
    for (int kt = 0; kt < KT; ++kt) {
      union { v16h v; _Float16 h[16]; } a;
      const _Float16* ap = A + m * aStride + kt * 32 + hi * 16;
      #pragma unroll
      for (int i = 0; i < 8; ++i) { a.h[i] = ap[i]; a.h[8 + i] = ap[8 + i]; }
      v16h b = *(const v16h*)(Wp + ((size_t)(kt * NT + nt) * 32 + lane) * 16);
      acc = __builtin_amdgcn_wmma_f32_16x16x32_f16(
          false, a.v, false, b, (short)0, acc, false, false);
    }
    const int n = nt * 16 + m;              // output column for this lane
    float bv = (n < fout) ? bias[n] : 0.f;  // padded cols: W=0 -> acc=0 -> store 0
    #pragma unroll
    for (int r = 0; r < 8; ++r) {
      float v = acc[r] + bv;
      if (act) v = selu_f(v);
      if (n < storeLimit) Out[(hi * 8 + r) * oStride + n] = (_Float16)v;
    }
  }
}

// ---------------------------------------------------------------------------
// Weight pack: f32 (fin x fout, row-major) -> f16 fragments in B layout.
// One 32-thread block per (kt,nt) fragment.
// ---------------------------------------------------------------------------
__global__ void pack_weight(const float* __restrict__ W, _Float16* __restrict__ dst,
                            int fin, int fout, int NT) {
  const int frag = blockIdx.x;
  const int kt = frag / NT, nt = frag % NT;
  const int lane = threadIdx.x;
  const int n   = nt * 16 + (lane & 15);
  const int klo = kt * 32 + (lane >> 4) * 16;
  _Float16* out = dst + ((size_t)frag * 32 + lane) * 16;
  #pragma unroll
  for (int i = 0; i < 16; ++i) {
    int k = klo + i;
    float v = (k < fin && n < fout) ? W[(size_t)k * fout + n] : 0.f;
    out[i] = (_Float16)v;
  }
}

__global__ void zero_u32(unsigned int* p, size_t n) {
  size_t i = (size_t)blockIdx.x * blockDim.x + threadIdx.x;
  size_t stride = (size_t)gridDim.x * blockDim.x;
  for (; i < n; i += stride) p[i] = 0u;
}

// ---------------------------------------------------------------------------
// Edge pipeline: permutate_nodes MLP + 3x update_edges MLP. 16 edges / wave.
// ---------------------------------------------------------------------------
struct EdgeP {
  const float *nodes, *edges;
  const int *snd, *rcv;
  const _Float16 *w0, *w1, *w2, *u0, *u1, *u2;
  const float *b0, *b1, *b2, *ub0, *ub1, *ub2;
  _Float16* h1e;
};

#define EW 8
__global__ __launch_bounds__(32 * EW) void edge_kernel(EdgeP p) {
  __shared__ _Float16 sA[EW][16 * 64];
  __shared__ _Float16 sB[EW][16 * 64];
  __shared__ _Float16 sH[EW][16 * 32];   // [0..4]=h1_edges [5..30]=h1_nodes [31]=0
  const int w = threadIdx.x >> 5, lane = threadIdx.x & 31;
  const int m = lane & 15, hi = lane >> 4;
  const long tile = (long)blockIdx.x * EW + w;
  const long e0 = tile * 16;
  if (e0 >= N_EDGES) return;             // whole-wave uniform exit
  _Float16 *A = sA[w], *B = sB[w], *H = sH[w];

  __builtin_prefetch(p.w0, 0, 3);
  __builtin_prefetch(p.u0, 0, 3);

  // gather: A row m = [nodes[rcv] | nodes[snd]] (26 cols, pad to 32)
  {
    long e = e0 + m;
    bool valid = e < N_EDGES;
    int src = valid ? ((hi == 0) ? p.rcv[e] : p.snd[e]) : 0;
    #pragma unroll
    for (int j = 0; j < ND; ++j) {
      float v = valid ? p.nodes[(size_t)src * ND + j] : 0.f;
      A[m * 64 + hi * ND + j] = (_Float16)v;
    }
    if (hi == 0) {
      for (int j = 2 * ND; j < 32; ++j) A[m * 64 + j] = (_Float16)0.f;
      #pragma unroll
      for (int j = 0; j < ED; ++j) {
        float v = valid ? p.edges[(size_t)e * ED + j] : 0.f;
        H[m * 32 + j] = (_Float16)v;
      }
      H[m * 32 + 31] = (_Float16)0.f;
    }
  }
  asm volatile("" ::: "memory");   // keep LDS fill ordered before WMMA reads

  // permutate_nodes: 26 -> 64 -> 64 -> 26
  gemm_tile(A, 64, p.w0, p.b0, 64, 1, 4, B, 64, 64, true);
  gemm_tile(B, 64, p.w1, p.b1, 64, 2, 4, A, 64, 64, true);
  gemm_tile(A, 64, p.w2, p.b2, 26, 2, 2, H + 5, 32, 26, false);

  // update_edges x3: concat(h1_edges[5], h1_nodes[26]) = 31 -> 64 -> 31 -> 5
  for (int it = 0; it < 3; ++it) {
    gemm_tile(H, 32, p.u0, p.ub0, 64, 1, 4, B, 64, 64, true);
    gemm_tile(B, 64, p.u1, p.ub1, 31, 2, 2, A, 64, 32, true);
    gemm_tile(A, 64, p.u2, p.ub2, 5, 1, 1, H, 32, 5, false);
  }

  if (hi == 0) {
    long e = e0 + m;
    if (e < N_EDGES)
      #pragma unroll
      for (int j = 0; j < ED; ++j) p.h1e[(size_t)e * ED + j] = H[m * 32 + j];
  }
}

// ---------------------------------------------------------------------------
// Scatter pass (sender-first ordering achieved by launching sender pass first)
// ---------------------------------------------------------------------------
__global__ void scatter_pass(const int* __restrict__ owner, const int* __restrict__ other,
                             const float* __restrict__ nodes, const _Float16* __restrict__ h1e,
                             _Float16* __restrict__ neigh, _Float16* __restrict__ efeat,
                             int* __restrict__ cnt) {
  long e = (long)blockIdx.x * blockDim.x + threadIdx.x;
  if (e >= N_EDGES) return;
  int o = owner[e], nb = other[e];
  int slot = atomicAdd(&cnt[o], 1);
  if (slot < KNB) {
    _Float16* nd = neigh + (size_t)o * (KNB * ND) + slot * ND;
    #pragma unroll
    for (int j = 0; j < ND; ++j) nd[j] = (_Float16)nodes[(size_t)nb * ND + j];
    _Float16* ed = efeat + (size_t)o * (KNB * ED) + slot * ED;
    #pragma unroll
    for (int j = 0; j < ED; ++j) ed[j] = h1e[(size_t)e * ED + j];
  }
}

// ---------------------------------------------------------------------------
// Node pipeline: permutate_neighbours + permutate_edges + 3x update_nodes +
// global-sum reduction. 16 nodes / wave, 4 waves / block.
// ---------------------------------------------------------------------------
struct NodeP {
  const float* nodes;
  const _Float16 *neigh, *efeat;
  const _Float16 *n0, *n1, *n2, *e0, *e1, *e2, *g0, *g1, *g2;
  const float *nb0, *nb1, *nb2, *eb0, *eb1, *eb2, *gb0, *gb1, *gb2;
  float* glob;
};

#define NW 4
__global__ __launch_bounds__(32 * NW) void node_kernel(NodeP p) {
  __shared__ _Float16 sN[NW][16 * 64];
  __shared__ _Float16 sB[NW][16 * 64];
  __shared__ _Float16 sE[NW][16 * 32];
  __shared__ _Float16 sH[NW][16 * 96];  // [0..12]=h2_nodes [13..64]=h_neigh [65..84]=h2_edges [85..95]=0
  __shared__ float gAcc[85];
  const int w = threadIdx.x >> 5, lane = threadIdx.x & 31;
  const int m = lane & 15, hi = lane >> 4;
  for (int i = threadIdx.x; i < 85; i += blockDim.x) gAcc[i] = 0.f;
  __syncthreads();

  const long tile = (long)blockIdx.x * NW + w;
  const long v0 = tile * 16;
  if (v0 < N_NODES) {
    _Float16 *A = sN[w], *B = sB[w], *Ebuf = sE[w], *H = sH[w];
    long v = v0 + m;
    bool valid = v < N_NODES;
    __builtin_prefetch(p.g0, 0, 3);

    if (hi == 0) {
      for (int j = 0; j < 52; ++j)
        A[m * 64 + j] = valid ? p.neigh[(size_t)v * 52 + j] : (_Float16)0.f;
      for (int j = 52; j < 64; ++j) A[m * 64 + j] = (_Float16)0.f;
    } else {
      for (int j = 0; j < 20; ++j)
        Ebuf[m * 32 + j] = valid ? p.efeat[(size_t)v * 20 + j] : (_Float16)0.f;
      for (int j = 20; j < 32; ++j) Ebuf[m * 32 + j] = (_Float16)0.f;
      #pragma unroll
      for (int j = 0; j < ND; ++j)
        H[m * 96 + j] = valid ? (_Float16)p.nodes[(size_t)v * ND + j] : (_Float16)0.f;
      for (int j = 85; j < 96; ++j) H[m * 96 + j] = (_Float16)0.f;
    }
    asm volatile("" ::: "memory");

    // permutate_neighbours: 52 -> 64 -> 64 -> 52  (into H cols 13..64)
    gemm_tile(A, 64, p.n0, p.nb0, 64, 2, 4, B, 64, 64, true);
    gemm_tile(B, 64, p.n1, p.nb1, 64, 2, 4, A, 64, 64, true);
    gemm_tile(A, 64, p.n2, p.nb2, 52, 2, 4, H + 13, 96, 52, false);

    // permutate_edges: 20 -> 64 -> 64 -> 20  (into H cols 65..84)
    gemm_tile(Ebuf, 32, p.e0, p.eb0, 64, 1, 4, B, 64, 64, true);
    gemm_tile(B, 64, p.e1, p.eb1, 64, 2, 4, A, 64, 64, true);
    gemm_tile(A, 64, p.e2, p.eb2, 20, 2, 2, H + 65, 96, 20, false);

    // update_nodes x3: 85 -> 50 -> 25 -> 13 (back into H cols 0..12)
    for (int it = 0; it < 3; ++it) {
      gemm_tile(H, 96, p.g0, p.gb0, 50, 3, 4, B, 64, 64, true);
      gemm_tile(B, 64, p.g1, p.gb1, 25, 2, 2, Ebuf, 32, 32, true);
      gemm_tile(Ebuf, 32, p.g2, p.gb2, 13, 1, 1, H, 96, 13, false);
    }

    // global aggregation: sum over the 16 rows of H cols 0..84
    for (int c = lane; c < 85; c += 32) {
      float s = 0.f;
      for (int r = 0; r < 16; ++r) {
        if (v0 + r < N_NODES) s += (float)H[r * 96 + c];
      }
      atomicAdd(&gAcc[c], s);                 // ds_add_f32
    }
  }
  __syncthreads();
  for (int i = threadIdx.x; i < 85; i += blockDim.x)
    atomicAdd(&p.glob[i], gAcc[i]);           // global_atomic_add_f32
}

// ---------------------------------------------------------------------------
// Predict head: [1,85] -> 50 -> 25 -> 12 (scalar, one block)
// ---------------------------------------------------------------------------
__global__ void predict_kernel(const float* __restrict__ glob,
                               const float* W0, const float* b0,
                               const float* W1, const float* b1,
                               const float* W2, const float* b2,
                               float* out) {
  __shared__ float g[85], h1[50], h2[25];
  int t = threadIdx.x;
  for (int i = t; i < 85; i += blockDim.x) g[i] = glob[i];
  __syncthreads();
  if (t < 50) {
    float s = b0[t];
    for (int k = 0; k < 85; ++k) s += g[k] * W0[k * 50 + t];
    h1[t] = selu_f(s);
  }
  __syncthreads();
  if (t < 25) {
    float s = b1[t];
    for (int k = 0; k < 50; ++k) s += h1[k] * W1[k * 25 + t];
    h2[t] = selu_f(s);
  }
  __syncthreads();
  if (t < 12) {
    float s = b2[t];
    for (int k = 0; k < 25; ++k) s += h2[k] * W2[k * 12 + t];
    out[t] = s;
  }
}

// ---------------------------------------------------------------------------
extern "C" void kernel_launch(void* const* d_in, const int* in_sizes, int n_in,
                              void* d_out, int out_size, void* d_ws, size_t ws_size,
                              hipStream_t stream) {
  (void)in_sizes; (void)n_in; (void)out_size; (void)ws_size;
  char* ws = (char*)d_ws;
  size_t off = 0;
  auto alloc = [&](size_t bytes) -> char* {
    char* p = ws + off;
    off = (off + bytes + 255) & ~(size_t)255;
    return p;
  };

  // --- pack all WMMA weights (f32 -> f16 B-fragment layout) ---
  struct WSpec { int fin, fout, idx; };
  const WSpec specs[15] = {
    {26, 64, 4}, {64, 64, 6}, {64, 26, 8},        // permutate_nodes
    {31, 64, 10}, {64, 31, 12}, {31, 5, 14},      // update_edges
    {52, 64, 16}, {64, 64, 18}, {64, 52, 20},     // permutate_neighbours
    {20, 64, 22}, {64, 64, 24}, {64, 20, 26},     // permutate_edges
    {85, 50, 28}, {50, 25, 30}, {25, 13, 32},     // update_nodes
  };
  _Float16* pk[15];
  for (int i = 0; i < 15; ++i) {
    int KT = (specs[i].fin + 31) / 32, NT = (specs[i].fout + 15) / 16;
    pk[i] = (_Float16*)alloc((size_t)KT * NT * 1024);
    pack_weight<<<dim3(KT * NT), dim3(32), 0, stream>>>(
        (const float*)d_in[specs[i].idx], pk[i], specs[i].fin, specs[i].fout, NT);
  }

  // --- scratch buffers ---
  _Float16* h1e   = (_Float16*)alloc((size_t)N_EDGES * ED * 2);
  _Float16* neigh = (_Float16*)alloc((size_t)N_NODES * (KNB * ND) * 2);
  _Float16* efeat = (_Float16*)alloc((size_t)N_NODES * (KNB * ED) * 2);
  int*      cnt   = (int*)alloc((size_t)N_NODES * 4);
  float*    glob  = (float*)alloc(85 * 4);

  zero_u32<<<4096, 256, 0, stream>>>((unsigned int*)neigh, (size_t)N_NODES * (KNB * ND) / 2);
  zero_u32<<<4096, 256, 0, stream>>>((unsigned int*)efeat, (size_t)N_NODES * (KNB * ED) / 2);
  zero_u32<<<2048, 256, 0, stream>>>((unsigned int*)cnt, (size_t)N_NODES);
  zero_u32<<<1, 128, 0, stream>>>((unsigned int*)glob, 85);

  const float* nodes = (const float*)d_in[0];
  const float* edges = (const float*)d_in[1];
  const int*   snd   = (const int*)d_in[2];
  const int*   rcv   = (const int*)d_in[3];

  // --- edge pipeline ---
  EdgeP ep;
  ep.nodes = nodes; ep.edges = edges; ep.snd = snd; ep.rcv = rcv;
  ep.w0 = pk[0]; ep.w1 = pk[1]; ep.w2 = pk[2];
  ep.u0 = pk[3]; ep.u1 = pk[4]; ep.u2 = pk[5];
  ep.b0  = (const float*)d_in[5];  ep.b1  = (const float*)d_in[7];  ep.b2  = (const float*)d_in[9];
  ep.ub0 = (const float*)d_in[11]; ep.ub1 = (const float*)d_in[13]; ep.ub2 = (const float*)d_in[15];
  ep.h1e = h1e;
  {
    unsigned tiles = (N_EDGES + 15) / 16;
    unsigned grid  = (tiles + EW - 1) / EW;
    edge_kernel<<<dim3(grid), dim3(32 * EW), 0, stream>>>(ep);
  }

  // --- collect neighbours & edges (sender pass first, then receiver pass) ---
  {
    unsigned grid = (N_EDGES + 255) / 256;
    scatter_pass<<<dim3(grid), dim3(256), 0, stream>>>(snd, rcv, nodes, h1e, neigh, efeat, cnt);
    scatter_pass<<<dim3(grid), dim3(256), 0, stream>>>(rcv, snd, nodes, h1e, neigh, efeat, cnt);
  }

  // --- node pipeline ---
  NodeP np;
  np.nodes = nodes; np.neigh = neigh; np.efeat = efeat;
  np.n0 = pk[6];  np.n1 = pk[7];  np.n2 = pk[8];
  np.e0 = pk[9];  np.e1 = pk[10]; np.e2 = pk[11];
  np.g0 = pk[12]; np.g1 = pk[13]; np.g2 = pk[14];
  np.nb0 = (const float*)d_in[17]; np.nb1 = (const float*)d_in[19]; np.nb2 = (const float*)d_in[21];
  np.eb0 = (const float*)d_in[23]; np.eb1 = (const float*)d_in[25]; np.eb2 = (const float*)d_in[27];
  np.gb0 = (const float*)d_in[29]; np.gb1 = (const float*)d_in[31]; np.gb2 = (const float*)d_in[33];
  np.glob = glob;
  {
    unsigned tiles = (N_NODES + 15) / 16;
    unsigned grid  = (tiles + NW - 1) / NW;
    node_kernel<<<dim3(grid), dim3(32 * NW), 0, stream>>>(np);
  }

  // --- predict head ---
  predict_kernel<<<dim3(1), dim3(128), 0, stream>>>(
      glob,
      (const float*)d_in[34], (const float*)d_in[35],
      (const float*)d_in[36], (const float*)d_in[37],
      (const float*)d_in[38], (const float*)d_in[39],
      (float*)d_out);
}
